// GNNClassifier_66881230733495
// MI455X (gfx1250) — matrix operand order repaired and checked by reading, not compiled, not run on hardware
//
#include <hip/hip_runtime.h>
#include <hip/hip_bf16.h>

// ---------------------------------------------------------------------------
// GNN classifier for MI455X (gfx1250, wave32, WMMA).
// GEMMs use v_wmma_f32_16x16x32_bf16; aggregation uses f32 global atomics
// (entire activation buffers are L2-resident: 2 x 51MB << 192MB L2).
// ---------------------------------------------------------------------------

typedef __attribute__((ext_vector_type(16))) __bf16 v16bf;
typedef __attribute__((ext_vector_type(8)))  float  v8f;

#define NODES  100000
#define EDGES  3200000
#define NGRAPH 64
#define INDIM  260
#define HID    128
#define NCLS   16

// ---------------------------------------------------------------------------
// Degree / normalization: deg = 1 (self loop) + in-degree; dinv = rsqrt(deg)
// ---------------------------------------------------------------------------
__global__ void deg_init(float* __restrict__ deg) {
    int i = blockIdx.x * blockDim.x + threadIdx.x;
    if (i < NODES) deg[i] = 1.0f;
}

__global__ void deg_accum(const int* __restrict__ ei, float* __restrict__ deg) {
    int e = blockIdx.x * blockDim.x + threadIdx.x;
    if (e < EDGES) atomicAdd(&deg[ei[EDGES + e]], 1.0f);
}

__global__ void deg_finish(float* __restrict__ deg) {
    int i = blockIdx.x * blockDim.x + threadIdx.x;
    if (i < NODES) deg[i] = rsqrtf(deg[i]);
}

// ---------------------------------------------------------------------------
// Prepack f32 weights (K x Ncols, row major) into the wave32 WMMA B-fragment
// layout (ISA 7.12.2, 16-bit B 32x16): fragment element j of lane L holds
// B[kBase + (L>>4)*8 + (j&7) + (j>>3)*16][tileN*16 + (L&15)], zero-padded past K.
// Buffer layout: [ks][tileN][lane][j]  -> each lane loads 32 contiguous bytes.
// ---------------------------------------------------------------------------
__global__ void pack_w(const float* __restrict__ W, __bf16* __restrict__ out,
                       int K, int Ncols, int kSteps) {
    int ntile = Ncols >> 4;
    int total = kSteps * ntile * 512;
    int idx = blockIdx.x * blockDim.x + threadIdx.x;
    if (idx >= total) return;
    int j    = idx & 15;
    int lane = (idx >> 4) & 31;
    int t    = idx >> 9;          // ks*ntile + tn
    int tn   = t % ntile;
    int ks   = t / ntile;
    int k    = ks * 32 + (lane >> 4) * 8 + (j & 7) + (j >> 3) * 16;
    int col  = tn * 16 + (lane & 15);
    out[idx] = (k < K) ? (__bf16)W[(size_t)k * Ncols + col] : (__bf16)0.0f;
}

// ---------------------------------------------------------------------------
// WMMA GEMM: C[M x Ncols] = A[M x K] (f32, cvt->bf16) @ Bpacked (bf16)
// One wave computes a 16-row x (NT*16)-col strip; f32 accumulation.
// A fragment per ISA 7.12.2 (16-bit A 16x32): lane (m,half) holds
// K runs [k0,k0+8) and [k0+16,k0+24) with k0 = ks*32 + half*8.
// The k-loop is split so the hot loop has ZERO divergence: a step is full
// iff (ks+1)*32 <= K, which is wave-uniform; at most one masked tail step.
// Optional fused bias / ReLU (used only for the MLP head).
// ---------------------------------------------------------------------------
template <int NT>
__global__ void __launch_bounds__(256)
gemm_bf16_wmma(const float* __restrict__ A, const __bf16* __restrict__ Bp,
               const float* __restrict__ bias, float* __restrict__ C,
               int M, int K, int Ncols, int relu) {
    int gid     = blockIdx.x * blockDim.x + threadIdx.x;
    int wave    = gid >> 5;
    int rowBase = wave << 4;
    if (rowBase >= M) return;                 // wave-uniform: EXEC stays all-1s
    int lane = threadIdx.x & 31;
    int m    = lane & 15;
    int half = lane >> 4;

    v8f acc[NT];
#pragma unroll
    for (int t = 0; t < NT; ++t) acc[t] = (v8f){0, 0, 0, 0, 0, 0, 0, 0};

    const float* arow = A + (size_t)(rowBase + m) * K;
    const int kFull = K >> 5;                 // complete 32-wide steps (uniform)

    for (int ks = 0; ks < kFull; ++ks) {
        int k0 = ks * 32 + half * 8;
        int k1 = k0 + 16;
        float4 x0 = *(const float4*)(arow + k0);
        float4 x1 = *(const float4*)(arow + k0 + 4);
        float4 y0 = *(const float4*)(arow + k1);
        float4 y1 = *(const float4*)(arow + k1 + 4);
        v16bf a;
        a[0]  = (__bf16)x0.x; a[1]  = (__bf16)x0.y;
        a[2]  = (__bf16)x0.z; a[3]  = (__bf16)x0.w;
        a[4]  = (__bf16)x1.x; a[5]  = (__bf16)x1.y;
        a[6]  = (__bf16)x1.z; a[7]  = (__bf16)x1.w;
        a[8]  = (__bf16)y0.x; a[9]  = (__bf16)y0.y;
        a[10] = (__bf16)y0.z; a[11] = (__bf16)y0.w;
        a[12] = (__bf16)y1.x; a[13] = (__bf16)y1.y;
        a[14] = (__bf16)y1.z; a[15] = (__bf16)y1.w;

        const __bf16* bbase = Bp + (((size_t)ks * NT) * 32 + lane) * 16;
#pragma unroll
        for (int t = 0; t < NT; ++t) {
            v16bf b = *(const v16bf*)(bbase + (size_t)t * 512);
            acc[t] = __builtin_amdgcn_wmma_f32_16x16x32_bf16(
                false, a, false, b, (short)0, acc[t], false, false);
        }
    }

    if (K & 31) {                             // single masked tail step
        int ks = kFull;
        int k0 = ks * 32 + half * 8;
        int k1 = k0 + 16;
        v16bf a;
#pragma unroll
        for (int i = 0; i < 8; ++i) {
            a[i]     = (k0 + i < K) ? (__bf16)arow[k0 + i] : (__bf16)0.0f;
            a[8 + i] = (k1 + i < K) ? (__bf16)arow[k1 + i] : (__bf16)0.0f;
        }
        const __bf16* bbase = Bp + (((size_t)ks * NT) * 32 + lane) * 16;
#pragma unroll
        for (int t = 0; t < NT; ++t) {
            v16bf b = *(const v16bf*)(bbase + (size_t)t * 512);
            acc[t] = __builtin_amdgcn_wmma_f32_16x16x32_bf16(
                false, a, false, b, (short)0, acc[t], false, false);
        }
    }

    // C/D layout: lane 0-15 -> N=lane, M=r ; lane 16-31 -> N=lane-16, M=8+r
#pragma unroll
    for (int t = 0; t < NT; ++t) {
        int col  = t * 16 + m;
        float bv = bias ? bias[col] : 0.0f;
#pragma unroll
        for (int r = 0; r < 8; ++r) {
            int row = rowBase + half * 8 + r;
            float v = acc[t][r] + bv;
            if (relu) v = fmaxf(v, 0.0f);
            if (row < M) C[(size_t)row * Ncols + col] = v;
        }
    }
}

// ---------------------------------------------------------------------------
// acc[i] = t[i] * dinv[i]^2   (self-loop term == accumulator initialization)
// one thread per float4 (node = idx>>5)
// ---------------------------------------------------------------------------
__global__ void self_init(const float* __restrict__ t, const float* __restrict__ dinv,
                          float* __restrict__ acc) {
    int idx = blockIdx.x * blockDim.x + threadIdx.x;
    if (idx >= NODES * (HID / 4)) return;
    int node = idx >> 5;
    float d = dinv[node];
    float s = d * d;
    float4 v = ((const float4*)t)[idx];
    float4 o;
    o.x = v.x * s; o.y = v.y * s; o.z = v.z * s; o.w = v.w * s;
    ((float4*)acc)[idx] = o;
}

// ---------------------------------------------------------------------------
// Edge scatter: warp per edge, lane covers 4 channels; 4 f32 atomics/lane.
// acc[dst] += t[src] * dinv[src]*dinv[dst]   (L2-resident atomics)
// ---------------------------------------------------------------------------
__global__ void scatter_edges(const int* __restrict__ ei, const float* __restrict__ dinv,
                              const float* __restrict__ t, float* __restrict__ acc) {
    int gid = blockIdx.x * blockDim.x + threadIdx.x;
    int e = gid >> 5;
    if (e >= EDGES) return;
    int lane = threadIdx.x & 31;
    int s = ei[e];
    int d = ei[EDGES + e];
    float w = dinv[s] * dinv[d];
    float4 v = ((const float4*)(t + (size_t)s * HID))[lane];
    float* dp = acc + (size_t)d * HID + lane * 4;
    atomicAdd(dp + 0, v.x * w);
    atomicAdd(dp + 1, v.y * w);
    atomicAdd(dp + 2, v.z * w);
    atomicAdd(dp + 3, v.w * w);
}

// ---------------------------------------------------------------------------
// In-place bias + ReLU (float4-vectorized)
// ---------------------------------------------------------------------------
__global__ void bias_relu(float* __restrict__ acc, const float* __restrict__ b) {
    int idx = blockIdx.x * blockDim.x + threadIdx.x;
    if (idx >= NODES * (HID / 4)) return;
    float4 v = ((const float4*)acc)[idx];
    float4 bb = ((const float4*)b)[idx & 31];
    float4 o;
    o.x = fmaxf(v.x + bb.x, 0.0f);
    o.y = fmaxf(v.y + bb.y, 0.0f);
    o.z = fmaxf(v.z + bb.z, 0.0f);
    o.w = fmaxf(v.w + bb.w, 0.0f);
    ((float4*)acc)[idx] = o;
}

// ---------------------------------------------------------------------------
// Global mean pool
// ---------------------------------------------------------------------------
__global__ void pool_zero(float* __restrict__ sums, float* __restrict__ cnts) {
    int idx = blockIdx.x * blockDim.x + threadIdx.x;
    if (idx < NGRAPH * HID) sums[idx] = 0.0f;
    if (idx < NGRAPH) cnts[idx] = 0.0f;
}

__global__ void pool_accum(const float* __restrict__ h, const int* __restrict__ batch,
                           float* __restrict__ sums, float* __restrict__ cnts) {
    int gid = blockIdx.x * blockDim.x + threadIdx.x;
    if (gid >= NODES * (HID / 4)) return;
    int node = gid >> 5;
    int lane = gid & 31;
    int g = batch[node];
    float4 v = ((const float4*)h)[gid];
    float* sp = sums + (size_t)g * HID + lane * 4;
    atomicAdd(sp + 0, v.x);
    atomicAdd(sp + 1, v.y);
    atomicAdd(sp + 2, v.z);
    atomicAdd(sp + 3, v.w);
    if (lane == 0) atomicAdd(&cnts[g], 1.0f);
}

__global__ void pool_div(const float* __restrict__ sums, const float* __restrict__ cnts,
                         float* __restrict__ pooled) {
    int idx = blockIdx.x * blockDim.x + threadIdx.x;
    if (idx >= NGRAPH * HID) return;
    int g = idx >> 7;
    pooled[idx] = sums[idx] / fmaxf(cnts[g], 1.0f);
}

// ---------------------------------------------------------------------------
// Host-side orchestration
// ---------------------------------------------------------------------------
extern "C" void kernel_launch(void* const* d_in, const int* in_sizes, int n_in,
                              void* d_out, int out_size, void* d_ws, size_t ws_size,
                              hipStream_t stream) {
    const float* x     = (const float*)d_in[0];
    const int*   ei    = (const int*)d_in[1];
    const int*   batch = (const int*)d_in[2];
    const float* w1    = (const float*)d_in[3];
    const float* b1    = (const float*)d_in[4];
    const float* w2    = (const float*)d_in[5];
    const float* b2    = (const float*)d_in[6];
    const float* w3    = (const float*)d_in[7];
    const float* b3    = (const float*)d_in[8];
    const float* fc1w  = (const float*)d_in[9];
    const float* fc1b  = (const float*)d_in[10];
    const float* fc2w  = (const float*)d_in[11];
    const float* fc2b  = (const float*)d_in[12];
    float* out = (float*)d_out;

    char* ws = (char*)d_ws;
    size_t off = 0;
    auto alloc = [&](size_t bytes) -> void* {
        void* p = ws + off;
        off += (bytes + 255) & ~((size_t)255);
        return p;
    };

    float*  dinv  = (float*)alloc(sizeof(float) * NODES);
    float*  bufA  = (float*)alloc(sizeof(float) * (size_t)NODES * HID);
    float*  bufB  = (float*)alloc(sizeof(float) * (size_t)NODES * HID);
    __bf16* w1p   = (__bf16*)alloc(sizeof(__bf16) * 9 * 8 * 512);  // K=260 -> 9 ksteps
    __bf16* w2p   = (__bf16*)alloc(sizeof(__bf16) * 4 * 8 * 512);  // K=128 -> 4 ksteps
    __bf16* w3p   = (__bf16*)alloc(sizeof(__bf16) * 4 * 8 * 512);
    __bf16* fc1p  = (__bf16*)alloc(sizeof(__bf16) * 4 * 8 * 512);
    __bf16* fc2p  = (__bf16*)alloc(sizeof(__bf16) * 4 * 1 * 512);
    float*  sums  = (float*)alloc(sizeof(float) * NGRAPH * HID);
    float*  cnts  = (float*)alloc(sizeof(float) * NGRAPH);
    float*  pooled= (float*)alloc(sizeof(float) * NGRAPH * HID);
    float*  z     = (float*)alloc(sizeof(float) * NGRAPH * HID);
    (void)ws_size; (void)in_sizes; (void)n_in; (void)out_size;

    const int BS = 256;
    auto cdiv = [](long long a, long long b) { return (int)((a + b - 1) / b); };

    // normalization coefficients
    deg_init  <<<cdiv(NODES, BS), BS, 0, stream>>>(dinv);
    deg_accum <<<cdiv(EDGES, BS), BS, 0, stream>>>(ei, dinv);
    deg_finish<<<cdiv(NODES, BS), BS, 0, stream>>>(dinv);

    // weight prepack (f32 -> bf16 WMMA B-fragment layout)
    pack_w<<<cdiv(9 * 8 * 512, BS), BS, 0, stream>>>(w1,   w1p, INDIM, HID, 9);
    pack_w<<<cdiv(4 * 8 * 512, BS), BS, 0, stream>>>(w2,   w2p, HID,   HID, 4);
    pack_w<<<cdiv(4 * 8 * 512, BS), BS, 0, stream>>>(w3,   w3p, HID,   HID, 4);
    pack_w<<<cdiv(4 * 8 * 512, BS), BS, 0, stream>>>(fc1w, fc1p, HID,  HID, 4);
    pack_w<<<cdiv(4 * 1 * 512, BS), BS, 0, stream>>>(fc2w, fc2p, HID, NCLS, 4);

    const int gemmBlocksN = cdiv((long long)(NODES / 16) * 32, BS);
    const int q4Blocks    = cdiv((long long)NODES * (HID / 4), BS);
    const int edgeBlocks  = cdiv((long long)EDGES * 32, BS);

    // ---- GCN layer 1: t = x @ w1 ; acc = selfloop + scatter ; h = relu(acc+b)
    gemm_bf16_wmma<8><<<gemmBlocksN, BS, 0, stream>>>(x, w1p, nullptr, bufA,
                                                      NODES, INDIM, HID, 0);
    self_init    <<<q4Blocks,   BS, 0, stream>>>(bufA, dinv, bufB);
    scatter_edges<<<edgeBlocks, BS, 0, stream>>>(ei, dinv, bufA, bufB);
    bias_relu    <<<q4Blocks,   BS, 0, stream>>>(bufB, b1);

    // ---- GCN layer 2
    gemm_bf16_wmma<8><<<gemmBlocksN, BS, 0, stream>>>(bufB, w2p, nullptr, bufA,
                                                      NODES, HID, HID, 0);
    self_init    <<<q4Blocks,   BS, 0, stream>>>(bufA, dinv, bufB);
    scatter_edges<<<edgeBlocks, BS, 0, stream>>>(ei, dinv, bufA, bufB);
    bias_relu    <<<q4Blocks,   BS, 0, stream>>>(bufB, b2);

    // ---- GCN layer 3
    gemm_bf16_wmma<8><<<gemmBlocksN, BS, 0, stream>>>(bufB, w3p, nullptr, bufA,
                                                      NODES, HID, HID, 0);
    self_init    <<<q4Blocks,   BS, 0, stream>>>(bufA, dinv, bufB);
    scatter_edges<<<edgeBlocks, BS, 0, stream>>>(ei, dinv, bufA, bufB);
    bias_relu    <<<q4Blocks,   BS, 0, stream>>>(bufB, b3);

    // ---- global mean pool
    pool_zero <<<cdiv(NGRAPH * HID, BS), BS, 0, stream>>>(sums, cnts);
    pool_accum<<<q4Blocks, BS, 0, stream>>>(bufB, batch, sums, cnts);
    pool_div  <<<cdiv(NGRAPH * HID, BS), BS, 0, stream>>>(sums, cnts, pooled);

    // ---- MLP head (fused bias+relu in GEMM)
    gemm_bf16_wmma<8><<<1, 128, 0, stream>>>(pooled, fc1p, fc1b, z,
                                             NGRAPH, HID, HID, 1);
    gemm_bf16_wmma<1><<<1, 128, 0, stream>>>(z, fc2p, fc2b, out,
                                             NGRAPH, HID, NCLS, 0);
}
